// MaskedBlock_55894704390250
// MI455X (gfx1250) — matrix operand order repaired and checked by reading, not compiled, not run on hardware
//
#include <hip/hip_runtime.h>
#include <cmath>

typedef __bf16 bf16_t;
typedef __attribute__((ext_vector_type(16))) __bf16 v16bf;
typedef __attribute__((ext_vector_type(8)))  __bf16 v8bf;
typedef __attribute__((ext_vector_type(4)))  __bf16 v4bf;
typedef __attribute__((ext_vector_type(8)))  float  v8f;

#define DIMC   768
#define C3     2304
#define HIDC   3072
#define NHEAD  12

__device__ inline v8bf zero8bf() {
  v8bf r;
#pragma unroll
  for (int i = 0; i < 8; i++) r[i] = (bf16_t)0.0f;
  return r;
}

// ---------------------------------------------------------------------------
// fp32 -> bf16 converter (weights)
// ---------------------------------------------------------------------------
__global__ void cvt_f32_bf16(const float* __restrict__ in, bf16_t* __restrict__ out, long n) {
  long i = (long)blockIdx.x * 256 + threadIdx.x;
  long stride = (long)gridDim.x * 256;
  for (; i < n; i += stride) out[i] = (bf16_t)in[i];
}

// ---------------------------------------------------------------------------
// LayerNorm over a 768-wide row, one wave per row, bf16 output
// ---------------------------------------------------------------------------
__device__ inline void ln_row(const float* __restrict__ src, const float* __restrict__ gw,
                              const float* __restrict__ bw, bf16_t* __restrict__ dst, int lane) {
  float v[24];
  float s = 0.f;
#pragma unroll
  for (int i = 0; i < 6; i++) {
    float4 t = *(const float4*)(src + i * 128 + lane * 4);
    v[4 * i + 0] = t.x; v[4 * i + 1] = t.y; v[4 * i + 2] = t.z; v[4 * i + 3] = t.w;
    s += t.x + t.y + t.z + t.w;
  }
#pragma unroll
  for (int o = 16; o > 0; o >>= 1) s += __shfl_xor(s, o, 32);
  float mu = s * (1.f / 768.f);
  float q = 0.f;
#pragma unroll
  for (int i = 0; i < 24; i++) { float d = v[i] - mu; q += d * d; }
#pragma unroll
  for (int o = 16; o > 0; o >>= 1) q += __shfl_xor(q, o, 32);
  float rstd = rsqrtf(q * (1.f / 768.f) + 1e-5f);
#pragma unroll
  for (int i = 0; i < 6; i++) {
    v4bf o4;
#pragma unroll
    for (int j = 0; j < 4; j++) {
      int c = i * 128 + lane * 4 + j;
      o4[j] = (bf16_t)((v[4 * i + j] - mu) * rstd * gw[c] + bw[c]);
    }
    *(v4bf*)(dst + i * 128 + lane * 4) = o4;
  }
}

__global__ __launch_bounds__(32) void ln_kernel(const float* __restrict__ x,
                                                const float* __restrict__ gw,
                                                const float* __restrict__ bw,
                                                bf16_t* __restrict__ out, int rows, int rpb,
                                                long bstride, int roff) {
  long r = blockIdx.x;
  if (r >= rows) return;
  long b = r / rpb, rr = r % rpb;
  ln_row(x + b * bstride + (rr + roff) * (long)DIMC, gw, bw, out + r * (long)DIMC, threadIdx.x);
}

// builds LN(xs) directly: row r = bt*197 + i with bt = b*16 + t
__global__ __launch_bounds__(32) void ln_spatial_kernel(const float* __restrict__ x,
                                                        const float* __restrict__ xt2,
                                                        const float* __restrict__ gw,
                                                        const float* __restrict__ bw,
                                                        bf16_t* __restrict__ out) {
  long r = blockIdx.x;            // 0..25215
  long bt = r / 197, i = r % 197;
  long b = bt / 16, t = bt % 16;
  const float* src = (i == 0) ? (x + b * 3137L * DIMC)
                              : (xt2 + (b * 3136L + (i - 1) * 16 + t) * DIMC);
  ln_row(src, gw, bw, out + r * (long)DIMC, threadIdx.x);
}

// ---------------------------------------------------------------------------
// WMMA GEMM: Y[M,N] = act(X[M,K] * W[N,K]^T + bias) (+ resid)   (bf16 in, f32 acc)
// Block tile 128x128, 8 waves (4x2), each wave 32x64 = 2x4 WMMA tiles, K slab 32.
// Loads use clamped row indices (no divergent zero-fill in the hot loop);
// out-of-range rows/cols only pollute D rows/cols that the epilogue never stores.
// ---------------------------------------------------------------------------
#define LDSK 40   // padded LDS row stride (elements): conflict-free 16-lane reads

template <int ACT, bool RESID, bool OUTBF>
__global__ __launch_bounds__(256) void gemm_wmma(const bf16_t* __restrict__ A,
                                                 const bf16_t* __restrict__ W,
                                                 const float* __restrict__ bias,
                                                 const float* __restrict__ resid,
                                                 void* __restrict__ outp, int M, int N, int K,
                                                 int r_rpb, long r_bstride, int r_roff) {
  __shared__ alignas(16) bf16_t sA[128 * LDSK];
  __shared__ alignas(16) bf16_t sB[128 * LDSK];
  const int tid = threadIdx.x;
  const int lane = tid & 31;
  const int wave = tid >> 5;
  const int wm = wave >> 1, wn = wave & 1;       // 4 waves along M, 2 along N
  const int nl = lane & 15, g = lane >> 4;
  const long bm = (long)blockIdx.x * 128;
  const long bn = (long)blockIdx.y * 128;

  v8f acc[2][4];
#pragma unroll
  for (int i = 0; i < 2; i++)
#pragma unroll
    for (int j = 0; j < 4; j++)
#pragma unroll
      for (int e = 0; e < 8; e++) acc[i][j][e] = 0.f;

  // Cooperative tile loads: 128 rows x 32 cols, 2 threads per row, 16 elems each
  const int arow = tid >> 1, acol = (tid & 1) * 16;
  long ga = bm + arow; if (ga > M - 1) ga = M - 1;   // clamp: no branches in loop
  long gb = bn + arow; if (gb > N - 1) gb = N - 1;
  const bf16_t* pa = A + ga * K + acol;
  const bf16_t* pw = W + gb * K + acol;

  for (int k0 = 0; k0 < K; k0 += 32) {
    v8bf a0 = *(const v8bf*)(pa + k0);
    v8bf a1 = *(const v8bf*)(pa + k0 + 8);
    v8bf b0 = *(const v8bf*)(pw + k0);
    v8bf b1 = *(const v8bf*)(pw + k0 + 8);
    __builtin_prefetch(pa + k0 + 32, 0, 1);   // speculative; safe past the end
    __builtin_prefetch(pw + k0 + 32, 0, 1);
    *(v8bf*)(&sA[arow * LDSK + acol]) = a0;
    *(v8bf*)(&sA[arow * LDSK + acol + 8]) = a1;
    *(v8bf*)(&sB[arow * LDSK + acol]) = b0;
    *(v8bf*)(&sB[arow * LDSK + acol + 8]) = b1;
    __syncthreads();

    v16bf af[2], bfm[4];
#pragma unroll
    for (int t = 0; t < 2; t++) {
      // A fragment: lane=row, K halves split per lane-group (CDNA5 16-bit A layout)
      const bf16_t* p = &sA[(wm * 32 + t * 16 + nl) * LDSK + g * 8];
      v8bf lo = *(const v8bf*)(p);
      v8bf hi = *(const v8bf*)(p + 16);
#pragma unroll
      for (int e = 0; e < 8; e++) { af[t][e] = lo[e]; af[t][e + 8] = hi[e]; }
    }
#pragma unroll
    for (int t = 0; t < 4; t++) {
      // B fragment: lane=col, K halves split per lane-group (same as A)
      const bf16_t* q = &sB[(wn * 64 + t * 16 + nl) * LDSK + g * 8];
      v8bf l2 = *(const v8bf*)(q);
      v8bf h2 = *(const v8bf*)(q + 16);
#pragma unroll
      for (int e = 0; e < 8; e++) { bfm[t][e] = l2[e]; bfm[t][e + 8] = h2[e]; }
    }
#pragma unroll
    for (int tm = 0; tm < 2; tm++)
#pragma unroll
      for (int tn = 0; tn < 4; tn++)
        acc[tm][tn] = __builtin_amdgcn_wmma_f32_16x16x32_bf16(
            false, af[tm], false, bfm[tn], (short)0, acc[tm][tn], false, false);
    __syncthreads();
  }

  // Epilogue: C/D layout -> lane holds column nl, rows g*8+e
#pragma unroll
  for (int tm = 0; tm < 2; tm++)
#pragma unroll
    for (int tn = 0; tn < 4; tn++) {
      long nn = bn + wn * 64 + tn * 16 + nl;
      if (nn >= N) continue;
      float bv = bias[nn];
#pragma unroll
      for (int e = 0; e < 8; e++) {
        long mm = bm + wm * 32 + tm * 16 + g * 8 + e;
        if (mm >= M) continue;
        float v = acc[tm][tn][e] + bv;
        if (ACT == 1) v = 0.5f * v * (1.f + erff(v * 0.70710678118654752f));
        if (RESID) {
          long rb = mm / r_rpb;
          long rr = mm % r_rpb;
          v += resid[rb * r_bstride + (rr + (long)r_roff) * N + nn];
        }
        if (OUTBF) ((bf16_t*)outp)[mm * N + nn] = (bf16_t)v;
        else       ((float*)outp)[mm * N + nn] = v;
      }
    }
}

// ---------------------------------------------------------------------------
// Flash attention, one wave per (batch, head, q-tile). d_head = 64.
// qkv rows: (bx*S + tok) x 2304 (q|k|v), bf16. Output (bx*S+tok) x 768 bf16.
// HAS_MASK: scores[q_row] *= (1 - suppress*mask[bx, q_row])  (temporal path)
// ---------------------------------------------------------------------------
template <int S, bool HAS_MASK>
__global__ __launch_bounds__(32) void attn_flash(const bf16_t* __restrict__ qkv,
                                                 const float* __restrict__ mask,
                                                 const float* __restrict__ suppress,
                                                 bf16_t* __restrict__ o, int BX) {
  constexpr int QT = (S + 15) / 16;
  const int bh = blockIdx.x / QT;
  const int qt = blockIdx.x % QT;
  const int bx = bh / NHEAD, h = bh % NHEAD;
  const int lane = threadIdx.x;
  const int nl = lane & 15, g = lane >> 4;

  __shared__ alignas(16) bf16_t sp[16 * 24];   // probabilities tile (row-major, stride 24)
  __shared__ alignas(16) bf16_t sv[16 * 72];   // V tile (16 keys x 64 d, stride 72)

  const bf16_t* qb = qkv + (long)bx * S * C3 + h * 64;
  const bf16_t* kb = qb + 768;
  const bf16_t* vb = qb + 1536;

  // Q fragments for K-dim chunks [0,32) and [32,64)
  v16bf aq[2];
  {
    int m = qt * 16 + nl;
    bool valid = (m < S);
    const bf16_t* qr = qb + (long)m * C3;
#pragma unroll
    for (int kc = 0; kc < 2; kc++) {
      v8bf lo = zero8bf(), hi = zero8bf();
      if (valid) {
        lo = *(const v8bf*)(qr + kc * 32 + g * 8);
        hi = *(const v8bf*)(qr + kc * 32 + 16 + g * 8);
      }
#pragma unroll
      for (int e = 0; e < 8; e++) { aq[kc][e] = lo[e]; aq[kc][e + 8] = hi[e]; }
    }
  }

  float factor[8];
  if (HAS_MASK) {
    float sup = *suppress;
#pragma unroll
    for (int e = 0; e < 8; e++) {
      int m = qt * 16 + g * 8 + e;   // S==16 here
      factor[e] = 1.f - sup * mask[(long)bx * 16 + m];
    }
  }

  float rm[8], rs[8];
  v8f oacc[4];
#pragma unroll
  for (int e = 0; e < 8; e++) { rm[e] = -1e30f; rs[e] = 0.f; }
#pragma unroll
  for (int d = 0; d < 4; d++)
#pragma unroll
    for (int e = 0; e < 8; e++) oacc[d][e] = 0.f;

  for (int kt = 0; kt < QT; kt++) {
    // K^T fragments (B matrix: K-dim = d, N = key)
    v16bf bk[2];
    {
      int n = kt * 16 + nl;
      bool valid = (n < S);
      const bf16_t* kr = kb + (long)n * C3;
#pragma unroll
      for (int kc = 0; kc < 2; kc++) {
        v8bf lo = zero8bf(), hi = zero8bf();
        if (valid) {
          lo = *(const v8bf*)(kr + kc * 32 + g * 8);
          hi = *(const v8bf*)(kr + kc * 32 + 16 + g * 8);
        }
#pragma unroll
        for (int e = 0; e < 8; e++) { bk[kc][e] = lo[e]; bk[kc][e + 8] = hi[e]; }
      }
    }
    v8f sc;
#pragma unroll
    for (int e = 0; e < 8; e++) sc[e] = 0.f;
    sc = __builtin_amdgcn_wmma_f32_16x16x32_bf16(false, aq[0], false, bk[0], (short)0, sc, false, false);
    sc = __builtin_amdgcn_wmma_f32_16x16x32_bf16(false, aq[1], false, bk[1], (short)0, sc, false, false);

    const bool kvalid = (kt * 16 + nl) < S;
    float p[8], corr[8];
#pragma unroll
    for (int e = 0; e < 8; e++) {
      float s = sc[e] * 0.125f;                 // 64^-0.5
      if (HAS_MASK) s *= factor[e];
      if (!kvalid) s = -1e30f;
      float tmax = s;
#pragma unroll
      for (int off = 1; off < 16; off <<= 1) tmax = fmaxf(tmax, __shfl_xor(tmax, off, 16));
      float mn = fmaxf(rm[e], tmax);
      float cr = __expf(rm[e] - mn);
      float pe = __expf(s - mn);
      float ps = pe;
#pragma unroll
      for (int off = 1; off < 16; off <<= 1) ps += __shfl_xor(ps, off, 16);
      rs[e] = rs[e] * cr + ps;
      rm[e] = mn;
      corr[e] = cr;
      p[e] = pe;
    }
#pragma unroll
    for (int d = 0; d < 4; d++)
#pragma unroll
      for (int e = 0; e < 8; e++) oacc[d][e] *= corr[e];

    // stage P (D-layout -> row-major LDS) and the V tile
#pragma unroll
    for (int e = 0; e < 8; e++) sp[(g * 8 + e) * 24 + nl] = (bf16_t)p[e];
    {
      int n = kt * 16 + nl;
      bool valid = (n < S);
      const bf16_t* vr = vb + (long)n * C3 + g * 32;
#pragma unroll
      for (int c = 0; c < 4; c++) {
        v8bf t = zero8bf();
        if (valid) t = *(const v8bf*)(vr + c * 8);
        *(v8bf*)(&sv[nl * 72 + g * 32 + c * 8]) = t;
      }
    }
    __syncthreads();

    // P as A fragment (16x32, upper K half zero)
    v16bf ap;
    {
      const bf16_t* pp = &sp[nl * 24 + g * 8];
      v8bf lo = *(const v8bf*)(pp);
#pragma unroll
      for (int e = 0; e < 8; e++) { ap[e] = lo[e]; ap[e + 8] = (bf16_t)0.f; }
    }
#pragma unroll
    for (int d = 0; d < 4; d++) {
      v16bf bv;
#pragma unroll
      for (int j = 0; j < 16; j++)
        bv[j] = (j < 8) ? sv[(g * 8 + j) * 72 + d * 16 + nl] : (bf16_t)0.f;
      oacc[d] = __builtin_amdgcn_wmma_f32_16x16x32_bf16(false, ap, false, bv, (short)0, oacc[d], false, false);
    }
    __syncthreads();
  }

#pragma unroll
  for (int e = 0; e < 8; e++) {
    int m = qt * 16 + g * 8 + e;
    if (m >= S) continue;
    float inv = (rs[e] > 0.f) ? (1.f / rs[e]) : 0.f;
#pragma unroll
    for (int d = 0; d < 4; d++) {
      int dd = d * 16 + nl;
      o[((long)bx * S + m) * (long)DIMC + h * 64 + dd] = (bf16_t)(oacc[d][e] * inv);
    }
  }
}

// ---------------------------------------------------------------------------
// out = concat([cls, xt2]) + concat([mean_t(res_s[:,0]), scatter(res_s[:,1:])])
// ---------------------------------------------------------------------------
__global__ void assemble_out_kernel(const float* __restrict__ x, const float* __restrict__ xt2,
                                    const float* __restrict__ res_s, float* __restrict__ out,
                                    long total) {
  long idx = (long)blockIdx.x * 256 + threadIdx.x;
  if (idx >= total) return;
  long c = idx % DIMC;
  long rem = idx / DIMC;
  long tok = rem % 3137, b = rem / 3137;
  float v;
  if (tok == 0) {
    float s = 0.f;
    for (int t = 0; t < 16; t++) s += res_s[((b * 16 + t) * 197L) * DIMC + c];
    v = x[(b * 3137L) * DIMC + c] + s * (1.f / 16.f);
  } else {
    long j = tok - 1;
    long hw = j >> 4, t = j & 15;
    v = xt2[(b * 3136L + j) * DIMC + c] + res_s[((b * 16 + t) * 197L + 1 + hw) * DIMC + c];
  }
  out[idx] = v;
}

// ---------------------------------------------------------------------------
extern "C" void kernel_launch(void* const* d_in, const int* in_sizes, int n_in,
                              void* d_out, int out_size, void* d_ws, size_t ws_size,
                              hipStream_t stream) {
  (void)in_sizes; (void)n_in; (void)out_size; (void)ws_size;
  const float* x        = (const float*)d_in[0];
  const float* mask     = (const float*)d_in[1];
  const float* Wqkv     = (const float*)d_in[2];
  const float* bqkv     = (const float*)d_in[3];
  const float* Wproj    = (const float*)d_in[4];
  const float* bproj    = (const float*)d_in[5];
  const float* tWqkv    = (const float*)d_in[6];
  const float* tbqkv    = (const float*)d_in[7];
  const float* tWproj   = (const float*)d_in[8];
  const float* tbproj   = (const float*)d_in[9];
  const float* suppress = (const float*)d_in[10];
  const float* tfcW     = (const float*)d_in[11];
  const float* tfcb     = (const float*)d_in[12];
  const float* g1       = (const float*)d_in[13];
  const float* b1       = (const float*)d_in[14];
  const float* tg1      = (const float*)d_in[15];
  const float* tb1      = (const float*)d_in[16];
  const float* g2       = (const float*)d_in[17];
  const float* b2       = (const float*)d_in[18];
  const float* fc1W     = (const float*)d_in[19];
  const float* fc1b     = (const float*)d_in[20];
  const float* fc2W     = (const float*)d_in[21];
  const float* fc2b     = (const float*)d_in[22];

  const int ROWS_T = 25088;  // 8*196*16
  const int ROWS_S = 25216;  // 128*197
  const int ROWS_O = 25096;  // 8*3137

  char* ws = (char*)d_ws;
  size_t off = 0;
  auto alloc = [&](size_t bytes) -> char* {
    char* p = ws + off;
    off += (bytes + 255) & ~(size_t)255;
    return p;
  };

  bf16_t* wWqkv  = (bf16_t*)alloc((size_t)C3 * DIMC * 2);
  bf16_t* wWproj = (bf16_t*)alloc((size_t)DIMC * DIMC * 2);
  bf16_t* wtWqkv = (bf16_t*)alloc((size_t)C3 * DIMC * 2);
  bf16_t* wtWproj= (bf16_t*)alloc((size_t)DIMC * DIMC * 2);
  bf16_t* wtfcW  = (bf16_t*)alloc((size_t)DIMC * DIMC * 2);
  bf16_t* wfc1W  = (bf16_t*)alloc((size_t)HIDC * DIMC * 2);
  bf16_t* wfc2W  = (bf16_t*)alloc((size_t)DIMC * HIDC * 2);
  bf16_t* R_A    = (bf16_t*)alloc((size_t)ROWS_S * DIMC * 2);           // LN outputs
  bf16_t* R_B    = (bf16_t*)alloc((size_t)ROWS_S * DIMC * 2);           // attention outputs
  bf16_t* R_C    = (bf16_t*)alloc((size_t)ROWS_T * DIMC * 2);           // proj_t output
  char*   R_BIG  = alloc((size_t)ROWS_O * HIDC * 2);                    // qkv / res_s / h1
  float*  R_XT2  = (float*)alloc((size_t)ROWS_T * DIMC * 4);
  float*  R_OUT  = (float*)alloc((size_t)ROWS_O * DIMC * 4);

  auto cvt = [&](const float* src, bf16_t* dst, long n) {
    int grid = (int)((n + 255) / 256);
    if (grid > 4096) grid = 4096;
    cvt_f32_bf16<<<grid, 256, 0, stream>>>(src, dst, n);
  };
  cvt(Wqkv,  wWqkv,  (long)C3 * DIMC);
  cvt(Wproj, wWproj, (long)DIMC * DIMC);
  cvt(tWqkv, wtWqkv, (long)C3 * DIMC);
  cvt(tWproj,wtWproj,(long)DIMC * DIMC);
  cvt(tfcW,  wtfcW,  (long)DIMC * DIMC);
  cvt(fc1W,  wfc1W,  (long)HIDC * DIMC);
  cvt(fc2W,  wfc2W,  (long)DIMC * HIDC);

  dim3 blk(256);

  // ---- temporal path ----
  ln_kernel<<<ROWS_T, 32, 0, stream>>>(x, tg1, tb1, R_A, ROWS_T, 3136, 3137L * DIMC, 1);

  bf16_t* qkv_t = (bf16_t*)R_BIG;
  gemm_wmma<0, false, true><<<dim3(196, 18), blk, 0, stream>>>(
      R_A, wtWqkv, tbqkv, nullptr, qkv_t, ROWS_T, C3, DIMC, 1, 0, 0);

  attn_flash<16, true><<<1568 * NHEAD, 32, 0, stream>>>(qkv_t, mask, suppress, R_B, 1568);

  gemm_wmma<0, false, true><<<dim3(196, 6), blk, 0, stream>>>(
      R_B, wtWproj, tbproj, nullptr, R_C, ROWS_T, DIMC, DIMC, 1, 0, 0);

  // xt2 = x[:,1:] + (attn_proj @ tfcW.T + tfcb)
  gemm_wmma<0, true, false><<<dim3(196, 6), blk, 0, stream>>>(
      R_C, wtfcW, tfcb, x, R_XT2, ROWS_T, DIMC, DIMC, 3136, 3137L * DIMC, 1);

  // ---- spatial path ----
  ln_spatial_kernel<<<ROWS_S, 32, 0, stream>>>(x, R_XT2, g1, b1, R_A);

  bf16_t* qkv_s = (bf16_t*)R_BIG;
  gemm_wmma<0, false, true><<<dim3(197, 18), blk, 0, stream>>>(
      R_A, wWqkv, bqkv, nullptr, qkv_s, ROWS_S, C3, DIMC, 1, 0, 0);

  attn_flash<197, false><<<128 * NHEAD * 13, 32, 0, stream>>>(qkv_s, nullptr, nullptr, R_B, 128);

  float* res_s = (float*)R_BIG;  // qkv_s is dead; reuse
  gemm_wmma<0, false, false><<<dim3(197, 6), blk, 0, stream>>>(
      R_B, wWproj, bproj, nullptr, res_s, ROWS_S, DIMC, DIMC, 1, 0, 0);

  long total = (long)ROWS_O * DIMC;
  assemble_out_kernel<<<(int)((total + 255) / 256), 256, 0, stream>>>(x, R_XT2, res_s, R_OUT, total);

  // ---- MLP ----
  ln_kernel<<<ROWS_O, 32, 0, stream>>>(R_OUT, g2, b2, R_A, ROWS_O, ROWS_O, 0, 0);

  bf16_t* h1 = (bf16_t*)R_BIG;  // res_s is dead; reuse
  gemm_wmma<1, false, true><<<dim3(197, 24), blk, 0, stream>>>(
      R_A, wfc1W, fc1b, nullptr, h1, ROWS_O, HIDC, DIMC, 1, 0, 0);

  gemm_wmma<0, true, false><<<dim3(197, 6), blk, 0, stream>>>(
      h1, wfc2W, fc2b, R_OUT, (float*)d_out, ROWS_O, DIMC, HIDC, ROWS_O, 0, 0);
}
